// SAE_3676492006104
// MI455X (gfx1250) — matrix operand order repaired and checked by reading, not compiled, not run on hardware
//
#include <hip/hip_runtime.h>
#include <hip/hip_bf16.h>

// Problem constants (match reference)
#define B_   8192
#define D_   768
#define W_   8192
#define K_   32

#define ROWS      32          // rows of x handled per workgroup (persistent)
#define DPAD      776         // 768 + 8 bf16 pad -> LDS row stride 388 dwords (=4 mod 64 banks)
#define NTHREADS  256         // 8 waves (wave32)
#define NWAVES    8
#define GROUPCOLS 64          // 4 N-tiles of 16 per pass
#define NGROUPS   (W_ / GROUPCOLS)           // 128
#define GROUPS_PER_WAVE (NGROUPS / NWAVES)   // 16

typedef __bf16 bf16_t;
typedef bf16_t v16bf __attribute__((ext_vector_type(16)));
typedef float  v8f   __attribute__((ext_vector_type(8)));
typedef float  v4f   __attribute__((ext_vector_type(4)));

union Frag { v16bf v; v4f f[2]; };

// ---------------- setup: column norms of Ad ([D,W], column j) ----------------
__global__ __launch_bounds__(NTHREADS) void sae_mp_norm(const float* __restrict__ Ad,
                                                        float* __restrict__ inv_norm) {
  int j = blockIdx.x * blockDim.x + threadIdx.x;   // coalesced over j
  float s = 0.f;
  for (int d = 0; d < D_; ++d) {
    float v = Ad[(size_t)d * W_ + j];
    s = fmaf(v, v, s);
  }
  inv_norm[j] = 1.0f / fmaxf(sqrtf(s), 1e-8f);
}

// ---- setup: AdnT[j][d] = Ad[d][j]/||col j||, fp32 + bf16 copies (transposed) ----
__global__ __launch_bounds__(NTHREADS) void sae_mp_build(const float* __restrict__ Ad,
                                                         const float* __restrict__ inv_norm,
                                                         float* __restrict__ AdnT32,
                                                         bf16_t* __restrict__ AdnT16) {
  int t = blockIdx.x * blockDim.x + threadIdx.x;   // t = j*D_ + d (writes coalesced)
  int j = t / D_;
  int d = t - j * D_;
  float v = Ad[(size_t)d * W_ + j] * inv_norm[j];
  AdnT32[t] = v;
  AdnT16[t] = (bf16_t)v;
}

// ---------------- main: persistent matching-pursuit kernel ----------------
__global__ __launch_bounds__(NTHREADS) void sae_mp_main(
    const float*  __restrict__ x,
    const float*  __restrict__ bd,
    const bf16_t* __restrict__ AdnT16,   // [W][D] bf16, L2-resident (12 MB)
    const float*  __restrict__ AdnT32,   // [W][D] f32,  L2-resident (25 MB)
    float*        __restrict__ resid32,  // [B][D] f32 workspace
    float*        __restrict__ out) {

  __shared__ bf16_t   s_r16[ROWS][DPAD];   // bf16 residual tile (A matrix)
  __shared__ unsigned s_maskC[W_];         // per-COLUMN word, bit m = row m selected
  __shared__ float    s_red_val[NWAVES][ROWS];
  __shared__ int      s_red_idx[NWAVES][ROWS];
  __shared__ float    s_winval[ROWS];
  __shared__ int      s_winidx[ROWS];

  const int tid  = threadIdx.x;
  const int wave = tid >> 5;
  const int lane = tid & 31;
  const int hh   = lane >> 4;   // half-wave (K-split per WMMA layout)
  const int mrow = lane & 15;   // row (A) / column (B) within tile
  const int row0 = blockIdx.x * ROWS;

  // resid0 = x - bd (fp32 to workspace, bf16 to LDS); clear selection mask
  for (int r = 0; r < ROWS; ++r) {
    const float* xr = x + (size_t)(row0 + r) * D_;
    float*       rr = resid32 + (size_t)(row0 + r) * D_;
    for (int d = tid; d < D_; d += NTHREADS) {
      float v = xr[d] - bd[d];
      rr[d] = v;
      s_r16[r][d] = (bf16_t)v;
    }
  }
  for (int i = tid; i < W_; i += NTHREADS) s_maskC[i] = 0u;
  __syncthreads();

  const char* bbase = (const char*)AdnT16;   // uniform scalar base -> saddr form

  for (int k = 0; k < K_; ++k) {
    // per-lane running argmax: [mt][r] covers row m = mt*16 + hh*8 + r
    // best magnitude is |best_val|; empty sentinel = (val 0, idx INT_MAX)
    float best_val[2][8];
    int   best_idx[2][8];
#pragma unroll
    for (int mt = 0; mt < 2; ++mt)
#pragma unroll
      for (int r = 0; r < 8; ++r) { best_val[mt][r] = 0.f; best_idx[mt][r] = 0x7FFFFFFF; }

    for (int gg = 0; gg < GROUPS_PER_WAVE; ++gg) {
      const int n0 = (wave + NWAVES * gg) * GROUPCOLS;

      v8f cacc[2][4];
#pragma unroll
      for (int mt = 0; mt < 2; ++mt)
#pragma unroll
        for (int tt = 0; tt < 4; ++tt)
#pragma unroll
          for (int e = 0; e < 8; ++e) cacc[mt][tt][e] = 0.f;

      // Per-lane 32-bit BYTE offset for B; the 4 tiles are reached with
      // compile-time immediate offsets (tt * 16 rows * 1536 B = 24576*tt).
      // Single v_add_nc_u32 per iteration, no 64-bit VALU address math ->
      // no WMMA->VALU WAR-hazard NOPs on the B operand registers.
      unsigned boff = (unsigned)((n0 + mrow) * D_ + hh * 8) * 2u;

      // NOTE: unroll 1 is essential — full unroll hoists 48 A-fragments into
      // registers, blows past 256 VGPRs and spills every A operand to scratch.
#pragma unroll 1
      for (int kk = 0; kk < D_; kk += 32) {
        // A fragments (both 16-row tiles) from LDS:
        // lane m holds row m; K = kk + hh*8 + [0..7] and kk + 16 + hh*8 + [0..7]
        Frag a0, a1;
        const bf16_t* pa0 = &s_r16[mrow][kk + hh * 8];
        const bf16_t* pa1 = &s_r16[16 + mrow][kk + hh * 8];
        a0.f[0] = *(const v4f*)pa0; a0.f[1] = *(const v4f*)(pa0 + 16);
        a1.f[0] = *(const v4f*)pa1; a1.f[1] = *(const v4f*)(pa1 + 16);

        // B fragments: lane n holds column n; K = kk + hh*8 + [0..7] and kk + 16 + hh*8 + [0..7]
        Frag bfr[4];
#pragma unroll
        for (int tt = 0; tt < 4; ++tt) {
          const char* pb = bbase + boff + (unsigned)(tt * 16 * D_ * 2);
          bfr[tt].f[0] = *(const v4f*)pb;
          bfr[tt].f[1] = *(const v4f*)(pb + 32);
        }
        boff += 64;   // 32 bf16 K-step

        // 8 independent WMMAs per K-step (hides XDL latency; B reused across M-tiles)
#pragma unroll
        for (int tt = 0; tt < 4; ++tt) {
          cacc[0][tt] = __builtin_amdgcn_wmma_f32_16x16x32_bf16(
              false, a0.v, false, bfr[tt].v, (short)0, cacc[0][tt], false, false);
          cacc[1][tt] = __builtin_amdgcn_wmma_f32_16x16x32_bf16(
              false, a1.v, false, bfr[tt].v, (short)0, cacc[1][tt], false, false);
        }
      }

      // masked running argmax over this group's 64 columns.
      // One broadcast mask word per tile; branchless cndmask updates so this
      // VALU tail co-executes with the next group's WMMA stream.
#pragma unroll
      for (int tt = 0; tt < 4; ++tt) {
        const int col = n0 + tt * 16 + mrow;
        const unsigned mw = s_maskC[col] >> (hh * 8);   // rows for this half at bits {mt*16+r}
#pragma unroll
        for (int mt = 0; mt < 2; ++mt)
#pragma unroll
          for (int r = 0; r < 8; ++r) {
            float z  = cacc[mt][tt][r];
            bool free_atom = ((mw >> (mt * 16 + r)) & 1u) == 0u;
            // strict > keeps lowest column (scanned ascending within a lane)
            bool take = free_atom & (__builtin_fabsf(z) > __builtin_fabsf(best_val[mt][r]));
            best_val[mt][r] = take ? z   : best_val[mt][r];
            best_idx[mt][r] = take ? col : best_idx[mt][r];
          }
      }
    }

    // cross-lane butterfly within 16-lane halves (each half owns 16 rows)
#pragma unroll
    for (int mt = 0; mt < 2; ++mt)
#pragma unroll
      for (int r = 0; r < 8; ++r) {
        float bv = best_val[mt][r];
        int   bi = best_idx[mt][r];
#pragma unroll
        for (int off = 1; off < 16; off <<= 1) {
          float ov = __shfl_xor(bv, off, 16);
          int   oi = __shfl_xor(bi, off, 16);
          float oa = __builtin_fabsf(ov), ba = __builtin_fabsf(bv);
          bool t = (oa > ba) | ((oa == ba) & (oi < bi));
          bv = t ? ov : bv;
          bi = t ? oi : bi;
        }
        if (mrow == 0) {
          int m = mt * 16 + hh * 8 + r;
          s_red_val[wave][m] = bv; s_red_idx[wave][m] = bi;
        }
      }
    __syncthreads();

    // cross-wave reduce: one thread per row picks winner, marks mask bit
    if (tid < ROWS) {
      float bv = s_red_val[0][tid];
      int   bi = s_red_idx[0][tid];
      for (int w = 1; w < NWAVES; ++w) {
        float ov = s_red_val[w][tid];
        int   oi = s_red_idx[w][tid];
        float oa = __builtin_fabsf(ov), ba = __builtin_fabsf(bv);
        bool t = (oa > ba) | ((oa == ba) & (oi < bi));
        bv = t ? ov : bv;
        bi = t ? oi : bi;
      }
      if (bi == 0x7FFFFFFF) { bi = 0; bv = 0.f; }   // degenerate all-masked guard
      s_winval[tid] = bv;
      s_winidx[tid] = bi;
      atomicOr(&s_maskC[bi], 1u << tid);            // ds_or_b32; rows may share a column
    }
    __syncthreads();

    // rank-1 fp32 residual update; refresh bf16 tile
    for (int r = 0; r < ROWS; ++r) {
      const float v    = s_winval[r];
      const float* ar  = AdnT32 + (size_t)s_winidx[r] * D_;
      float*       rr  = resid32 + (size_t)(row0 + r) * D_;
      for (int d = tid; d < D_; d += NTHREADS) {
        float nv = rr[d] - v * ar[d];
        rr[d] = nv;
        s_r16[r][d] = (bf16_t)nv;
      }
    }
    __syncthreads();
  }

  // out = codes @ Ad_n.T + bd == x - resid_final  (skips the reconstruction GEMM)
  for (int r = 0; r < ROWS; ++r) {
    const size_t off = (size_t)(row0 + r) * D_;
    for (int d = tid; d < D_; d += NTHREADS)
      out[off + d] = x[off + d] - resid32[off + d];
  }
}

// ---------------------------------------------------------------------------
extern "C" void kernel_launch(void* const* d_in, const int* in_sizes, int n_in,
                              void* d_out, int out_size, void* d_ws, size_t ws_size,
                              hipStream_t stream) {
  const float* x  = (const float*)d_in[0];   // [B, D]
  const float* Ad = (const float*)d_in[1];   // [D, W]
  const float* bd = (const float*)d_in[2];   // [1, D]
  float* out = (float*)d_out;                // [B, D]

  char* ws = (char*)d_ws;
  float*  AdnT32   = (float*)ws;   ws += (size_t)W_ * D_ * sizeof(float);
  bf16_t* AdnT16   = (bf16_t*)ws;  ws += (size_t)W_ * D_ * sizeof(bf16_t);
  float*  inv_norm = (float*)ws;   ws += (size_t)W_ * sizeof(float);
  float*  resid32  = (float*)ws;   // [B, D] fp32 residual

  sae_mp_norm <<<W_ / NTHREADS, NTHREADS, 0, stream>>>(Ad, inv_norm);
  sae_mp_build<<<(W_ * D_) / NTHREADS, NTHREADS, 0, stream>>>(Ad, inv_norm, AdnT32, AdnT16);
  sae_mp_main <<<B_ / ROWS, NTHREADS, 0, stream>>>(x, bd, AdnT16, AdnT32, resid32, out);
}
